// Model_13245679140924
// MI455X (gfx1250) — matrix operand order repaired and checked by reading, not compiled, not run on hardware
//
#include <hip/hip_runtime.h>
#include <hip/hip_bf16.h>

// ---------------------------------------------------------------------------
// Types for CDNA5 WMMA (wave32): v_wmma_f32_16x16x32_bf16
// ---------------------------------------------------------------------------
typedef __attribute__((ext_vector_type(16))) __bf16        v16bf;
typedef __attribute__((ext_vector_type(8)))  __bf16        v8bf;
typedef __attribute__((ext_vector_type(8)))  float         v8f;
typedef __attribute__((ext_vector_type(8)))  unsigned short v8us;
typedef __attribute__((ext_vector_type(16))) unsigned short v16us;

__device__ __forceinline__ unsigned short f32_to_bf16_rne(float f) {
  unsigned int u = __builtin_bit_cast(unsigned int, f);
  u += 0x7FFFu + ((u >> 16) & 1u);
  return (unsigned short)(u >> 16);
}

// ---------------------------------------------------------------------------
// Utility kernels
// ---------------------------------------------------------------------------
__global__ void k_zero(float* p, int n) {
  int i = blockIdx.x * 256 + threadIdx.x;
  if (i < n) p[i] = 0.f;
}

__global__ void k_f32_to_bf16(const float* __restrict__ s, unsigned short* __restrict__ d, int n) {
  int i = blockIdx.x * 256 + threadIdx.x;
  if (i < n) d[i] = f32_to_bf16_rne(s[i]);
}

// Transpose + convert + K-pad a weight matrix: src f32 [K, N] -> dst bf16 [N, Kpad]
__global__ void k_convT(const float* __restrict__ src, unsigned short* __restrict__ dst,
                        int K, int Kpad, int N) {
  int idx = blockIdx.x * 256 + threadIdx.x;
  if (idx >= N * Kpad) return;
  int c = idx / Kpad;
  int k = idx - c * Kpad;
  dst[idx] = (k < K) ? f32_to_bf16_rne(src[(size_t)k * N + c]) : (unsigned short)0;
}

__global__ void k_count(const int* __restrict__ idx, float* __restrict__ cnt, int E) {
  int e = blockIdx.x * 256 + threadIdx.x;
  if (e < E) atomicAdd(cnt + idx[e], 1.0f);
}

// per-edge scatter: hbuf[dst] += feat[src] * rsqrt(max(outdeg[src],1))
__global__ void k_scatter(const float* __restrict__ feat, int ldf,
                          const int* __restrict__ src, const int* __restrict__ dst,
                          const float* __restrict__ cnt_src,
                          float* __restrict__ hbuf, int ldh, int D) {
  int e = blockIdx.x;
  int s = src[e], d = dst[e];
  float sc = rsqrtf(fmaxf(cnt_src[s], 1.0f));
  const float* f = feat + (size_t)s * ldf;
  float* h = hbuf + (size_t)d * ldh;
  for (int j = threadIdx.x; j < D; j += blockDim.x)
    atomicAdd(h + j, f[j] * sc);
}

// write unclipped in-degree as extra feature column; rowscale = indeg^-0.5 (clipped)
__global__ void k_sethe(const float* __restrict__ cnt_dst, float* __restrict__ hbuf,
                        int ldh, int D, float* __restrict__ rowscale, int n) {
  int i = blockIdx.x * 256 + threadIdx.x;
  if (i < n) {
    float c = cnt_dst[i];
    hbuf[(size_t)i * ldh + D] = c;
    rowscale[i] = rsqrtf(fmaxf(c, 1.0f));
  }
}

// ---------------------------------------------------------------------------
// WMMA GEMM: C[M,N] = act( (A_bf16[M,Kpad] @ BT_bf16[N,Kpad]^T) * rowscale[M] + bias[N] )
//   - K multiple of 32 (buffers are zero-padded by construction) -> branchless loop
//   - B pre-transposed [N, Kpad]: every WMMA fragment (A and B) is two
//     contiguous global_load_b128 per lane; no LDS, no syncthreads
//   - wave32: each wave -> 16x64 C strip = 4x v_wmma_f32_16x16x32_bf16 per K-step
//   - grid = (N/64, M/64), block = 128 (4 waves)
// ---------------------------------------------------------------------------
__global__ __launch_bounds__(128) void k_gemm_wmma(
    const unsigned short* __restrict__ A, int lda,
    const unsigned short* __restrict__ BT, int ldbt,
    float* __restrict__ C, int ldc, int K,
    const float* __restrict__ rowscale,
    const float* __restrict__ bias,
    int act) {
  const int lane = threadIdx.x & 31;
  const int wave = threadIdx.x >> 5;
  const int row0 = blockIdx.y * 64 + wave * 16;
  const int col0 = blockIdx.x * 64;
  const int lrow = lane & 15;
  const int lhi  = lane >> 4;

  // A fragment: lane holds row (row0+lrow); elems j = K (k0 + lhi*16 + j)
  // (same contiguous K order as B)
  const unsigned short* Ap = A + (size_t)(row0 + lrow) * lda + lhi * 16;
  // B fragment: lane holds col (col0+nt*16+lrow); elems j = K (k0 + lhi*16 + j)
  const unsigned short* Bp = BT + (size_t)(col0 + lrow) * ldbt + lhi * 16;
  const size_t bstride = (size_t)16 * ldbt;  // advance 16 columns

  v8f acc[4] = {};

  for (int k0 = 0; k0 < K; k0 += 32) {
    v8bf alo = __builtin_bit_cast(v8bf, *(const v8us*)(Ap + k0));
    v8bf ahi = __builtin_bit_cast(v8bf, *(const v8us*)(Ap + k0 + 8));
    v16bf afrag = __builtin_shufflevector(alo, ahi,
        0, 1, 2, 3, 4, 5, 6, 7, 8, 9, 10, 11, 12, 13, 14, 15);
#pragma unroll
    for (int nt = 0; nt < 4; ++nt) {
      v16bf bfrag = __builtin_bit_cast(v16bf, *(const v16us*)(Bp + nt * bstride + k0));
      acc[nt] = __builtin_amdgcn_wmma_f32_16x16x32_bf16(
          false, afrag, false, bfrag, (short)0, acc[nt], false, false);
    }
  }

  // epilogue (C/D 16x16 f32 layout: lanes 0-15 -> M=vgpr, lanes 16-31 -> M=8+vgpr)
  const int mb = lhi * 8;
#pragma unroll
  for (int nt = 0; nt < 4; ++nt) {
    int col = col0 + nt * 16 + lrow;
#pragma unroll
    for (int i = 0; i < 8; ++i) {
      int r = row0 + mb + i;
      float v = acc[nt][i];
      if (rowscale) v *= rowscale[r];
      if (bias)     v += bias[col];
      if (act)      v = (v > 0.f) ? v : (__expf(v) - 1.f);  // ELU
      C[(size_t)r * ldc + col] = v;
    }
  }
}

// ---------------------------------------------------------------------------
// GRU gate math (Keras reset_after). xt = cx[b] + dx[t] (b_in folded into dx),
// rec = h@Wh + b_rec (bias folded into GEMM). Writes h both f32 (next gate)
// and bf16 (next recurrent WMMA GEMM A operand). One thread per (b, j<256).
// ---------------------------------------------------------------------------
__global__ void k_gru_gate(const float* __restrict__ cx, const float* __restrict__ dxt,
                           const float* __restrict__ rec, const float* __restrict__ hprev,
                           float* __restrict__ hnext, unsigned short* __restrict__ hnext_bf,
                           float* __restrict__ ystore, int ystride) {
  int idx = blockIdx.x * 256 + threadIdx.x;
  int b = idx >> 8;
  int j = idx & 255;
  const float* cxb = cx + (size_t)b * 768;
  const float* rb  = rec + (size_t)b * 768;
  float xz = cxb[j]       + dxt[j];
  float xr = cxb[256 + j] + dxt[256 + j];
  float xh = cxb[512 + j] + dxt[512 + j];
  float z  = 1.f / (1.f + __expf(-(xz + rb[j])));
  float r  = 1.f / (1.f + __expf(-(xr + rb[256 + j])));
  float hh = tanhf(xh + r * rb[512 + j]);
  float h  = hprev[idx];
  float hn = z * h + (1.f - z) * hh;
  hnext[idx] = hn;
  hnext_bf[idx] = f32_to_bf16_rne(hn);
  ystore[(size_t)b * ystride + j] = hn;
}

// ---------------------------------------------------------------------------
// Fused final projection + log_softmax for one time step (N=3, VALU path).
// ---------------------------------------------------------------------------
__global__ void k_final(const float* __restrict__ yft, int yfs,
                        const float* __restrict__ ybt, int ybs,
                        const float* __restrict__ Wf, const float* __restrict__ bfv,
                        float* __restrict__ outt, int outs) {
  int b = blockIdx.x * 256 + threadIdx.x;
  const float* yf = yft + (size_t)b * yfs;
  const float* yb = ybt + (size_t)b * ybs;
  float l0 = bfv[0], l1 = bfv[1], l2 = bfv[2];
  for (int j = 0; j < 256; ++j) {
    float v = yf[j];
    l0 = fmaf(v, Wf[3 * j], l0);
    l1 = fmaf(v, Wf[3 * j + 1], l1);
    l2 = fmaf(v, Wf[3 * j + 2], l2);
  }
  for (int j = 0; j < 256; ++j) {
    float v = yb[j];
    l0 = fmaf(v, Wf[3 * (256 + j)], l0);
    l1 = fmaf(v, Wf[3 * (256 + j) + 1], l1);
    l2 = fmaf(v, Wf[3 * (256 + j) + 2], l2);
  }
  float m  = fmaxf(l0, fmaxf(l1, l2));
  float e0 = __expf(l0 - m), e1 = __expf(l1 - m), e2 = __expf(l2 - m);
  float ls = __logf(e0 + e1 + e2);
  float* o = outt + (size_t)b * outs;
  o[0] = (l0 - m) - ls;
  o[1] = (l1 - m) - ls;
  o[2] = (l2 - m) - ls;
}

// ---------------------------------------------------------------------------
// Host orchestration
// ---------------------------------------------------------------------------
extern "C" __attribute__((visibility("default")))
void kernel_launch(void* const* d_in, const int* in_sizes, int n_in,
                   void* d_out, int out_size, void* d_ws, size_t ws_size,
                   hipStream_t stream) {
  (void)in_sizes; (void)n_in; (void)out_size; (void)ws_size;

  const float* comp_feat = (const float*)d_in[0];
  const float* dev_feat  = (const float*)d_in[1];
  const int*   comp_src  = (const int*)d_in[2];
  const int*   comp_dst  = (const int*)d_in[3];
  const int*   dev_src   = (const int*)d_in[4];
  const int*   dev_dst   = (const int*)d_in[5];
  const float* W1c  = (const float*)d_in[6];
  const float* b1c  = (const float*)d_in[7];
  const float* W2c  = (const float*)d_in[8];
  const float* b2c  = (const float*)d_in[9];
  const float* W1d  = (const float*)d_in[10];
  const float* b1d  = (const float*)d_in[11];
  const float* W2d  = (const float*)d_in[12];
  const float* b2d  = (const float*)d_in[13];
  const float* Wx_f = (const float*)d_in[14];
  const float* Wh_f = (const float*)d_in[15];
  const float* b_f  = (const float*)d_in[16];
  const float* Wx_b = (const float*)d_in[17];
  const float* Wh_b = (const float*)d_in[18];
  const float* b_b  = (const float*)d_in[19];
  const float* Wf   = (const float*)d_in[20];
  const float* bfv  = (const float*)d_in[21];
  float* out = (float*)d_out;

  // ---- workspace carve-out (256B aligned) ----
  char* wsb = (char*)d_ws;
  size_t off = 0;
  auto take = [&](size_t bytes) -> char* {
    char* p = wsb + off;
    off = (off + bytes + 255) & ~(size_t)255;
    return p;
  };
  // transposed, K-padded bf16 weights [N, Kpad]
  unsigned short* W1cT  = (unsigned short*)take((size_t)1024 * 160 * 2);   // K=129
  unsigned short* W2cT  = (unsigned short*)take((size_t)1024 * 1056 * 2);  // K=1025
  unsigned short* W1dT  = (unsigned short*)take((size_t)1024 * 64 * 2);    // K=33
  unsigned short* W2dT  = (unsigned short*)take((size_t)1024 * 1056 * 2);  // K=1025
  unsigned short* WxFtT = (unsigned short*)take((size_t)768 * 1024 * 2);   // Wx_f[:1024]
  unsigned short* WxFbT = (unsigned short*)take((size_t)768 * 1024 * 2);   // Wx_f[1024:]
  unsigned short* WxBtT = (unsigned short*)take((size_t)768 * 1024 * 2);
  unsigned short* WxBbT = (unsigned short*)take((size_t)768 * 1024 * 2);
  unsigned short* WhFT  = (unsigned short*)take((size_t)768 * 256 * 2);
  unsigned short* WhBT  = (unsigned short*)take((size_t)768 * 256 * 2);
  // graph buffers
  float* cntsC = (float*)take(1024 * 4);
  float* cntdC = (float*)take(1024 * 4);
  float* cntsD = (float*)take(64 * 4);
  float* cntdD = (float*)take(64 * 4);
  float* rsC   = (float*)take(1024 * 4);
  float* rsD   = (float*)take(64 * 4);
  float* hbufC          = (float*)take((size_t)1024 * 1056 * 4);
  unsigned short* hbufCb = (unsigned short*)take((size_t)1024 * 1056 * 2);
  float* hbufD          = (float*)take((size_t)64 * 1056 * 4);
  unsigned short* hbufDb = (unsigned short*)take((size_t)64 * 1056 * 2);
  float* x1C   = (float*)take((size_t)1024 * 1024 * 4);
  float* x1D   = (float*)take((size_t)64 * 1024 * 4);
  float* compEmb          = (float*)take((size_t)1024 * 1024 * 4);
  unsigned short* compEmbB = (unsigned short*)take((size_t)1024 * 1024 * 2);
  float* devEmb           = (float*)take((size_t)64 * 1024 * 4);
  unsigned short* devEmbB  = (unsigned short*)take((size_t)64 * 1024 * 2);
  // GRU buffers
  float* cxf = (float*)take((size_t)1024 * 768 * 4);
  float* cxb = (float*)take((size_t)1024 * 768 * 4);
  float* dxf = (float*)take((size_t)64 * 768 * 4);
  float* dxb = (float*)take((size_t)64 * 768 * 4);
  float* rec = (float*)take((size_t)1024 * 768 * 4);
  float* hA  = (float*)take((size_t)1024 * 256 * 4);
  float* hB  = (float*)take((size_t)1024 * 256 * 4);
  unsigned short* hAb = (unsigned short*)take((size_t)1024 * 256 * 2);
  unsigned short* hBb = (unsigned short*)take((size_t)1024 * 256 * 2);
  float* ybuf = (float*)take((size_t)1024 * 256 * 4);
  float* yf   = (float*)take((size_t)1024 * 64 * 256 * 4);

  auto zero = [&](float* p, int n) {
    k_zero<<<(n + 255) / 256, 256, 0, stream>>>(p, n);
  };
  auto tobf = [&](const float* s, unsigned short* d, int n) {
    k_f32_to_bf16<<<(n + 255) / 256, 256, 0, stream>>>(s, d, n);
  };
  auto convT = [&](const float* s, unsigned short* d, int K, int Kpad, int N) {
    int n = N * Kpad;
    k_convT<<<(n + 255) / 256, 256, 0, stream>>>(s, d, K, Kpad, N);
  };
  auto gemm = [&](const unsigned short* A, int lda, const unsigned short* BT, int ldbt,
                  float* Cc, int ldc, int M, int N, int K,
                  const float* rs, const float* bi, int act) {
    dim3 g(N / 64, M / 64);
    k_gemm_wmma<<<g, 128, 0, stream>>>(A, lda, BT, ldbt, Cc, ldc, K, rs, bi, act);
  };

  // ---- weight transpose + bf16 conversion (+ K zero-padding) ----
  convT(W1c, W1cT, 129, 160, 1024);
  convT(W2c, W2cT, 1025, 1056, 1024);
  convT(W1d, W1dT, 33, 64, 1024);
  convT(W2d, W2dT, 1025, 1056, 1024);
  convT(Wx_f,                      WxFtT, 1024, 1024, 768);
  convT(Wx_f + (size_t)1024 * 768, WxFbT, 1024, 1024, 768);
  convT(Wx_b,                      WxBtT, 1024, 1024, 768);
  convT(Wx_b + (size_t)1024 * 768, WxBbT, 1024, 1024, 768);
  convT(Wh_f, WhFT, 256, 256, 768);
  convT(Wh_b, WhBT, 256, 256, 768);

  // ---- GraphConv layers ----
  auto gconv = [&](const float* feat, int D, int Kpad, const int* src, const int* dst,
                   int n, int E, float* cnt_s, float* cnt_d, float* rowsc,
                   float* hbuf, unsigned short* hbufb,
                   const unsigned short* WT, const float* bi, float* outb, int act, bool deg) {
    if (deg) {
      zero(cnt_s, n);
      zero(cnt_d, n);
      k_count<<<(E + 255) / 256, 256, 0, stream>>>(src, cnt_s, E);
      k_count<<<(E + 255) / 256, 256, 0, stream>>>(dst, cnt_d, E);
    }
    zero(hbuf, n * Kpad);  // zeros pad columns too
    k_scatter<<<E, 128, 0, stream>>>(feat, D, src, dst, cnt_s, hbuf, Kpad, D);
    k_sethe<<<(n + 255) / 256, 256, 0, stream>>>(cnt_d, hbuf, Kpad, D, rowsc, n);
    tobf(hbuf, hbufb, n * Kpad);
    gemm(hbufb, Kpad, WT, Kpad, outb, 1024, n, 1024, Kpad, rowsc, bi, act);
  };

  gconv(comp_feat, 128, 160, comp_src, comp_dst, 1024, 16384,
        cntsC, cntdC, rsC, hbufC, hbufCb, W1cT, b1c, x1C, 1, true);
  gconv(x1C, 1024, 1056, comp_src, comp_dst, 1024, 16384,
        cntsC, cntdC, rsC, hbufC, hbufCb, W2cT, b2c, compEmb, 0, false);
  gconv(dev_feat, 32, 64, dev_src, dev_dst, 64, 1024,
        cntsD, cntdD, rsD, hbufD, hbufDb, W1dT, b1d, x1D, 1, true);
  gconv(x1D, 1024, 1056, dev_src, dev_dst, 64, 1024,
        cntsD, cntdD, rsD, hbufD, hbufDb, W2dT, b2d, devEmb, 0, false);

  tobf(compEmb, compEmbB, 1024 * 1024);
  tobf(devEmb, devEmbB, 64 * 1024);

  // ---- factored GRU input projections (broadcast-concat algebra) ----
  // x[b,t]@Wx = comp_emb[b]@Wx[:1024] + dev_emb[t]@Wx[1024:] (+ b_in, folded into dx)
  gemm(compEmbB, 1024, WxFtT, 1024, cxf, 768, 1024, 768, 1024, nullptr, nullptr, 0);
  gemm(compEmbB, 1024, WxBtT, 1024, cxb, 768, 1024, 768, 1024, nullptr, nullptr, 0);
  gemm(devEmbB, 1024, WxFbT, 1024, dxf, 768, 64, 768, 1024, nullptr, b_f, 0);
  gemm(devEmbB, 1024, WxBbT, 1024, dxb, 768, 64, 768, 1024, nullptr, b_b, 0);

  // ---- forward GRU: 64 sequential (recurrent WMMA GEMM + gates) steps ----
  zero(hA, 1024 * 256);
  zero((float*)hAb, 1024 * 256 / 2);
  float* hc = hA;  float* hn = hB;
  unsigned short* hcb = hAb;  unsigned short* hnb = hBb;
  for (int t = 0; t < 64; ++t) {
    gemm(hcb, 256, WhFT, 256, rec, 768, 1024, 768, 256, nullptr, b_f + 768, 0);
    k_gru_gate<<<1024, 256, 0, stream>>>(cxf, dxf + (size_t)t * 768, rec, hc, hn, hnb,
                                         yf + (size_t)t * 256, 64 * 256);
    float* tf = hc; hc = hn; hn = tf;
    unsigned short* tb = hcb; hcb = hnb; hnb = tb;
  }

  // ---- backward GRU fused with final projection + log_softmax ----
  zero(hA, 1024 * 256);
  zero((float*)hAb, 1024 * 256 / 2);
  hc = hA;  hn = hB;  hcb = hAb;  hnb = hBb;
  for (int tt = 0; tt < 64; ++tt) {
    int t = 63 - tt;
    gemm(hcb, 256, WhBT, 256, rec, 768, 1024, 768, 256, nullptr, b_b + 768, 0);
    k_gru_gate<<<1024, 256, 0, stream>>>(cxb, dxb + (size_t)t * 768, rec, hc, hn, hnb,
                                         ybuf, 256);
    k_final<<<1024 / 256, 256, 0, stream>>>(yf + (size_t)t * 256, 64 * 256, ybuf, 256,
                                            Wf, bfv, out + (size_t)t * 3, 64 * 3);
    float* tf = hc; hc = hn; hn = tf;
    unsigned short* tb = hcb; hcb = hnb; hnb = tb;
  }
}
